// SpaTrans_962072674925
// MI455X (gfx1250) — matrix operand order repaired and checked by reading, not compiled, not run on hardware
//
#include <hip/hip_runtime.h>
#include <hip/hip_bf16.h>

// ---------------------------------------------------------------------------
// Problem dims (fixed by the reference): b=1, c=64, a=25, h=w=32
//   L = 1024 tokens, Bm = 25 batches, E = 128, heads=8, Dh=16, K9=576, F=256
// ---------------------------------------------------------------------------

typedef __bf16 bf16_t;
typedef __attribute__((ext_vector_type(16))) __bf16 v16bf;
typedef __attribute__((ext_vector_type(8)))  __bf16 v8bf;
typedef __attribute__((ext_vector_type(8)))  float  v8f;

#define LTOK   1024
#define BM     25
#define EDIM   128
#define MTOT   (BM * LTOK)   // 25600
#define K9     576
#define NHEAD  8
#define DHEAD  16

__device__ __forceinline__ bf16_t f2bf(float f) { return (bf16_t)f; }

__device__ __forceinline__ v8f wmma_bf16(v16bf a, v16bf b, v8f c) {
  return __builtin_amdgcn_wmma_f32_16x16x32_bf16(
      false, a, false, b, (short)0, c, false, false);
}

// Fragment layouts (ISA 7.12.2 / 7.12.5):
//  A 16x32 (as loaded here): lane<16 row=lane k=0..15; lane>=16 row=lane-16 k=16..31
//  B 32x16 (as loaded here, same k order as A): lane<16 col=lane k=0..15; lane>=16 col=lane-16 k=16..31
//  C/D 16x16 f32: elem r -> row = r + (lane>=16?8:0), col = lane&15

// ---------------------------------------------------------------------------
// Weight pre-conversion f32 -> bf16 (weights are tiny; runs once per call)
// ---------------------------------------------------------------------------
__global__ void cvt_kernel(const float* __restrict__ in, bf16_t* __restrict__ out,
                           const int n) {
  const int i = blockIdx.x * 256 + threadIdx.x;
  if (i < n) out[i] = f2bf(in[i]);
}

// ---------------------------------------------------------------------------
// Kernel 1: token GEMM  out[n][l][e] = sum_k unfold3(src)[k] * w_mlp[e][k]
// Incremental (c, dy, dx) tracking: zero integer divides in the K loop.
// grid (64, 25), block 128 (4 waves x 2 col tiles)
// ---------------------------------------------------------------------------
__global__ void token_gemm_kernel(const float* __restrict__ src,
                                  const bf16_t* __restrict__ w_mlp,
                                  float* __restrict__ out) {
  __shared__ bf16_t Atile[16][40];

  const int tid  = threadIdx.x;
  const int lane = tid & 31;
  const int wave = tid >> 5;
  const int l0   = blockIdx.x * 16;
  const int n    = blockIdx.y;

  const int arow = lane & 15;
  const int kb8  = (lane >= 16) ? 8 : 0;
  const int kb16 = (lane >= 16) ? 16 : 0;
  const int n0   = wave * 32;
  const int n1   = n0 + 16;

  // staging coordinates for this thread
  const int r  = tid >> 3;          // 0..15
  const int kk = (tid & 7) * 4;     // 0,4,..,28
  const int l  = l0 + r;
  const int qi = l >> 5, qj = l & 31;
  const float* srcn = src + ((size_t)n << 10);  // + c*25*1024 later

  // decompose kk = c*9 + dy*3 + dx  (one-time divides at kernel entry)
  int c  = kk / 9;
  int rm = kk - 9 * c;
  int dy = rm / 3;
  int dx = rm - 3 * dy;

  v8f acc0 = {0.f,0.f,0.f,0.f,0.f,0.f,0.f,0.f};
  v8f acc1 = {0.f,0.f,0.f,0.f,0.f,0.f,0.f,0.f};

  for (int kbase = 0; kbase < K9; kbase += 32) {
    // ---- stage im2col A tile: 4 consecutive k per thread, incremental coords
    {
      int cu = c, dyu = dy, dxu = dx;
      #pragma unroll
      for (int u = 0; u < 4; ++u) {
        const int ii = qi + dyu - 1;
        const int jj = qj + dxu - 1;
        const bool inb = ((unsigned)ii < 32u) && ((unsigned)jj < 32u);
        const int iic = ii < 0 ? 0 : (ii > 31 ? 31 : ii);
        const int jjc = jj < 0 ? 0 : (jj > 31 ? 31 : jj);
        const float vv = srcn[((size_t)cu * (BM << 10)) + (iic << 5) + jjc];
        Atile[r][kk + u] = f2bf(inb ? vv : 0.f);
        if (++dxu == 3) { dxu = 0; if (++dyu == 3) { dyu = 0; ++cu; } }
      }
      // advance persistent coords by 32 = 3*9 + 1*3 + 2
      c += 3;
      dx += 2; if (dx >= 3) { dx -= 3; ++dy; }
      dy += 1; if (dy >= 3) { dy -= 3; ++c; }
    }
    __syncthreads();

    v16bf af;
    {
      const v8bf lo = *(const v8bf*)&Atile[arow][kb16];
      const v8bf hi = *(const v8bf*)&Atile[arow][kb16 + 8];
      #pragma unroll
      for (int e = 0; e < 8; ++e) { af[e] = lo[e]; af[8 + e] = hi[e]; }
    }

    const v16bf bf0 = *(const v16bf*)(w_mlp + (size_t)(n0 + (lane & 15)) * K9 + kbase + kb16);
    const v16bf bf1 = *(const v16bf*)(w_mlp + (size_t)(n1 + (lane & 15)) * K9 + kbase + kb16);

    acc0 = wmma_bf16(af, bf0, acc0);
    acc1 = wmma_bf16(af, bf1, acc1);
    __syncthreads();
  }

  const int mrow_off = (lane >= 16) ? 8 : 0;
  #pragma unroll
  for (int rr = 0; rr < 8; ++rr) {
    const size_t m = (size_t)n * LTOK + l0 + rr + mrow_off;
    out[m * EDIM + n0 + (lane & 15)] = acc0[rr];
    out[m * EDIM + n1 + (lane & 15)] = acc1[rr];
  }
}

// ---------------------------------------------------------------------------
// ln1:  x = LN(tok+pe)*g+b (bf16), tok_bf = bf16(tok).  8 waves/block, row/wave
// ---------------------------------------------------------------------------
__global__ void ln1_kernel(const float* __restrict__ tok, const float* __restrict__ pe,
                           const float* __restrict__ g, const float* __restrict__ b,
                           bf16_t* __restrict__ xout, bf16_t* __restrict__ tokout) {
  const int wave = threadIdx.x >> 5, lane = threadIdx.x & 31;
  const size_t row = (size_t)blockIdx.x * 8 + wave;
  const float* tr = tok + row * EDIM;
  const float* pr = pe  + row * EDIM;
  float tv[4], sv[4], sum = 0.f, sq = 0.f;
  #pragma unroll
  for (int q = 0; q < 4; ++q) {
    const int e = lane * 4 + q;
    tv[q] = tr[e]; sv[q] = tv[q] + pr[e];
    sum += sv[q]; sq += sv[q] * sv[q];
  }
  #pragma unroll
  for (int off = 16; off; off >>= 1) {
    sum += __shfl_xor(sum, off);
    sq  += __shfl_xor(sq, off);
  }
  const float mean = sum * (1.f / EDIM);
  const float var  = sq * (1.f / EDIM) - mean * mean;
  const float rstd = rsqrtf(var + 1e-5f);
  #pragma unroll
  for (int q = 0; q < 4; ++q) {
    const int e = lane * 4 + q;
    xout[row * EDIM + e]   = f2bf((sv[q] - mean) * rstd * g[e] + b[e]);
    tokout[row * EDIM + e] = f2bf(tv[q]);
  }
}

__global__ void ln2_kernel(const float* __restrict__ in,
                           const float* __restrict__ g, const float* __restrict__ b,
                           bf16_t* __restrict__ out) {
  const int wave = threadIdx.x >> 5, lane = threadIdx.x & 31;
  const size_t row = (size_t)blockIdx.x * 8 + wave;
  const float* ir = in + row * EDIM;
  float sv[4], sum = 0.f, sq = 0.f;
  #pragma unroll
  for (int q = 0; q < 4; ++q) {
    const int e = lane * 4 + q;
    sv[q] = ir[e]; sum += sv[q]; sq += sv[q] * sv[q];
  }
  #pragma unroll
  for (int off = 16; off; off >>= 1) {
    sum += __shfl_xor(sum, off);
    sq  += __shfl_xor(sq, off);
  }
  const float mean = sum * (1.f / EDIM);
  const float var  = sq * (1.f / EDIM) - mean * mean;
  const float rstd = rsqrtf(var + 1e-5f);
  #pragma unroll
  for (int q = 0; q < 4; ++q) {
    const int e = lane * 4 + q;
    out[row * EDIM + e] = f2bf((sv[q] - mean) * rstd * g[e] + b[e]);
  }
}

// ---------------------------------------------------------------------------
// Generic WMMA linear: out[m][nn] = sum_k in[m][k] * Wt[nn][k]
//   in bf16 [M][K], Wt bf16 [N][K].  64-row M-tile: each wave reuses its two
//   B fragments across 4 A fragments -> 8 wmma per K-step.
// grid (M/64, N/128), block 128 (4 waves x 2 col tiles x 4 row tiles)
// MODE (compile-time): 0 bf16 | 1 bf16 relu | 2 f32+residual | 3 bf16+residual
//                      | 4 f32 transposed out[nn*M+m]
// ---------------------------------------------------------------------------
template <int MODE>
__global__ void linear_kernel(const bf16_t* __restrict__ in, const bf16_t* __restrict__ Wt,
                              const int M, const int N, const int K,
                              bf16_t* __restrict__ out_bf, float* __restrict__ out_f,
                              const float* __restrict__ residual) {
  __shared__ bf16_t Atile[64][48];   // 96B row stride: 32B-aligned v16bf slots

  const int tid  = threadIdx.x;
  const int lane = tid & 31;
  const int wave = tid >> 5;
  const int m0   = blockIdx.x * 64;
  const int cb   = blockIdx.y * 128;

  const int arow = lane & 15;
  const int kb8  = (lane >= 16) ? 8 : 0;
  const int kb16 = (lane >= 16) ? 16 : 0;
  const int n0   = cb + wave * 32;
  const int n1   = n0 + 16;
  const bool do0 = (n0 < N), do1 = (n1 < N);

  const int sr  = tid >> 1;          // staging row 0..63
  const int skk = (tid & 1) * 16;    // staging k offset (32B granularity)

  v8f acc0[4], acc1[4];
  #pragma unroll
  for (int rt = 0; rt < 4; ++rt) {
    acc0[rt] = (v8f){0.f,0.f,0.f,0.f,0.f,0.f,0.f,0.f};
    acc1[rt] = (v8f){0.f,0.f,0.f,0.f,0.f,0.f,0.f,0.f};
  }

  for (int kbase = 0; kbase < K; kbase += 32) {
    *(v16bf*)&Atile[sr][skk] =
        *(const v16bf*)(in + (size_t)(m0 + sr) * K + kbase + skk);
    __syncthreads();

    v16bf bf0, bf1;
    if (do0) bf0 = *(const v16bf*)(Wt + (size_t)(n0 + (lane & 15)) * K + kbase + kb16);
    if (do1) bf1 = *(const v16bf*)(Wt + (size_t)(n1 + (lane & 15)) * K + kbase + kb16);

    #pragma unroll
    for (int rt = 0; rt < 4; ++rt) {
      v16bf af;
      const v8bf lo = *(const v8bf*)&Atile[rt * 16 + arow][kb16];
      const v8bf hi = *(const v8bf*)&Atile[rt * 16 + arow][kb16 + 8];
      #pragma unroll
      for (int e = 0; e < 8; ++e) { af[e] = lo[e]; af[8 + e] = hi[e]; }
      if (do0) acc0[rt] = wmma_bf16(af, bf0, acc0[rt]);
      if (do1) acc1[rt] = wmma_bf16(af, bf1, acc1[rt]);
    }
    __syncthreads();
  }

  const int mrow_off = (lane >= 16) ? 8 : 0;
  #pragma unroll
  for (int rt = 0; rt < 4; ++rt) {
    #pragma unroll
    for (int r = 0; r < 8; ++r) {
      const size_t m = (size_t)m0 + rt * 16 + r + mrow_off;
      #pragma unroll
      for (int t = 0; t < 2; ++t) {
        const bool dot = t ? do1 : do0;
        if (!dot) continue;
        const int nn = (t ? n1 : n0) + (lane & 15);
        float v = t ? acc1[rt][r] : acc0[rt][r];
        if (MODE == 1) v = fmaxf(v, 0.f);
        if (MODE == 2 || MODE == 3) v += residual[m * EDIM + nn];
        if (MODE == 2)      out_f[m * (size_t)N + nn] = v;
        else if (MODE == 4) out_f[(size_t)nn * M + m] = v;
        else                out_bf[m * (size_t)N + nn] = f2bf(v);
      }
    }
  }
}

// ---------------------------------------------------------------------------
// Local attention: one wave per (16-query tile, head, batch).
// 5x5 mask => only key rows qi-2..qi+2 (10 key tiles / 160 keys). Dh padded to 32.
// ---------------------------------------------------------------------------
__global__ void attn_kernel(const bf16_t* __restrict__ q, const bf16_t* __restrict__ k,
                            const bf16_t* __restrict__ v, bf16_t* __restrict__ out) {
  __shared__ float  sc[16][176];
  __shared__ bf16_t pb[16][160];

  const int lane = threadIdx.x;
  const int l0 = blockIdx.x * 16;
  const int h  = blockIdx.y;
  const int n  = blockIdx.z;
  const int qi  = l0 >> 5;
  const int qj0 = l0 & 31;

  const bf16_t* qp = q + ((size_t)n * LTOK) * EDIM + h * DHEAD;
  const bf16_t* kp = k + ((size_t)n * LTOK) * EDIM + h * DHEAD;
  const bf16_t* vp = v + ((size_t)n * LTOK) * EDIM + h * DHEAD;

  const int kb8  = (lane >= 16) ? 8 : 0;
  const int kb16 = (lane >= 16) ? 16 : 0;
  const int mrow_off = (lane >= 16) ? 8 : 0;

  // query A fragment (d padded 16->32 with zeros)
  v16bf qa;
  {
    const v8bf qv = *(const v8bf*)(qp + (size_t)(l0 + (lane & 15)) * EDIM + kb8);
    #pragma unroll
    for (int e = 0; e < 8; ++e) { qa[e] = qv[e]; qa[8 + e] = f2bf(0.0f); }
  }

  // ---- scores: 5 bands x 2 key tiles ----
  for (int band = 0; band < 5; ++band) {
    const int ki = qi - 2 + band;
    const bool valid = (ki >= 0) && (ki < 32);
    for (int half = 0; half < 2; ++half) {
      const int mbase = ki * 32 + half * 16;
      const int tile  = band * 2 + half;
      v8f acc = {0.f,0.f,0.f,0.f,0.f,0.f,0.f,0.f};
      if (valid) {
        v16bf kf;
        #pragma unroll
        for (int e = 0; e < 16; ++e) kf[e] = f2bf(0.0f);
        {   // half h holds d = 8h..8h+7 in e<8; e>=8 (d>=16) zero pad
          const v8bf kv = *(const v8bf*)(kp + (size_t)(mbase + (lane & 15)) * EDIM + kb8);
          #pragma unroll
          for (int e = 0; e < 8; ++e) kf[e] = kv[e];
        }
        acc = wmma_bf16(qa, kf, acc);
      }
      const int kj = half * 16 + (lane & 15);
      #pragma unroll
      for (int r = 0; r < 8; ++r) {
        const int qrow_m = r + mrow_off;
        const int qj = qj0 + qrow_m;
        const bool ok = valid && (kj >= qj - 2) && (kj < qj + 3);
        sc[qrow_m][tile * 16 + (lane & 15)] = ok ? acc[r] * 0.25f : -__builtin_inff();
      }
    }
  }

  // ---- softmax over 160 keys (lanes 0..15, one row each) ----
  if (lane < 16) {
    float mx = -__builtin_inff();
    for (int c2 = 0; c2 < 160; ++c2) mx = fmaxf(mx, sc[lane][c2]);
    float s = 0.f;
    for (int c2 = 0; c2 < 160; ++c2) {
      const float e2 = __expf(sc[lane][c2] - mx);
      sc[lane][c2] = e2; s += e2;
    }
    const float inv = 1.0f / s;
    for (int c2 = 0; c2 < 160; ++c2) pb[lane][c2] = f2bf(sc[lane][c2] * inv);
  }

  // ---- out = P(16x160) x V(160x16) ----
  v8f oacc = {0.f,0.f,0.f,0.f,0.f,0.f,0.f,0.f};
  for (int band = 0; band < 5; ++band) {
    const int ki = qi - 2 + band;
    const int cbase = band * 32;
    v16bf pa;
    {
      const v8bf lo = *(const v8bf*)&pb[lane & 15][cbase + kb16];
      const v8bf hi = *(const v8bf*)&pb[lane & 15][cbase + kb16 + 8];
      #pragma unroll
      for (int e = 0; e < 8; ++e) { pa[e] = lo[e]; pa[8 + e] = hi[e]; }
    }
    v16bf vf;
    if (ki >= 0 && ki < 32) {
      #pragma unroll
      for (int e = 0; e < 16; ++e) {
        const int kk = e + kb16;
        vf[e] = vp[(size_t)(ki * 32 + kk) * EDIM + (lane & 15)];
      }
    } else {
      #pragma unroll
      for (int e = 0; e < 16; ++e) vf[e] = f2bf(0.0f);  // probs are 0 anyway
    }
    oacc = wmma_bf16(pa, vf, oacc);
  }

  #pragma unroll
  for (int r = 0; r < 8; ++r) {
    const size_t l = (size_t)l0 + r + mrow_off;
    out[((size_t)n * LTOK + l) * EDIM + h * DHEAD + (lane & 15)] = f2bf(oacc[r]);
  }
}

// ---------------------------------------------------------------------------
// Host orchestration
// ---------------------------------------------------------------------------
extern "C" void kernel_launch(void* const* d_in, const int* in_sizes, int n_in,
                              void* d_out, int out_size, void* d_ws, size_t ws_size,
                              hipStream_t stream) {
  const float* buffer = (const float*)d_in[0];
  const float* spa    = (const float*)d_in[1];
  const float* w_mlp  = (const float*)d_in[2];   // [128][576]
  const float* ln1g   = (const float*)d_in[3];
  const float* ln1b   = (const float*)d_in[4];
  const float* inpw   = (const float*)d_in[5];   // [384][128]
  const float* outpw  = (const float*)d_in[6];   // [128][128]
  const float* ln2g   = (const float*)d_in[7];
  const float* ln2b   = (const float*)d_in[8];
  const float* ffw1   = (const float*)d_in[9];   // [256][128]
  const float* ffw2   = (const float*)d_in[10];  // [128][256]
  const float* convw  = (const float*)d_in[11];  // [64][128]
  float* out = (float*)d_out;                    // [64][25600]

  char* ws = (char*)d_ws;
  const size_t F32MAT = (size_t)MTOT * EDIM * sizeof(float);   // 13,107,200
  const size_t BFMAT  = (size_t)MTOT * EDIM * sizeof(bf16_t);  //  6,553,600
  float*  tokf  = (float*) (ws);                       // tok f32 (residual 1)
  float*  pef   = (float*) (ws + F32MAT);              // pe f32   -> tok2 f32
  bf16_t* xbf   = (bf16_t*)(ws + 2*F32MAT);            // LN1 out  -> attn out
  bf16_t* tokbf = (bf16_t*)(ws + 2*F32MAT +   BFMAT);  // tok bf16 -> ff_in
  bf16_t* qb    = (bf16_t*)(ws + 2*F32MAT + 2*BFMAT);  // q  -> ff_mid (q..k)
  bf16_t* kb    = (bf16_t*)(ws + 2*F32MAT + 3*BFMAT);  // k
  bf16_t* vb    = (bf16_t*)(ws + 2*F32MAT + 4*BFMAT);  // v  -> tok3 bf16

  // bf16 weight mirror
  bf16_t* w_mlp_bf = (bf16_t*)(ws + 2*F32MAT + 5*BFMAT);
  const int n_mlp = 128 * 576;
  bf16_t* inpw_bf  = w_mlp_bf + n_mlp;   const int n_inp = 384 * 128;
  bf16_t* outpw_bf = inpw_bf + n_inp;    const int n_out = 128 * 128;
  bf16_t* ffw1_bf  = outpw_bf + n_out;   const int n_ff1 = 256 * 128;
  bf16_t* ffw2_bf  = ffw1_bf + n_ff1;    const int n_ff2 = 128 * 256;
  bf16_t* convw_bf = ffw2_bf + n_ff2;    const int n_cnv = 64 * 128;

  cvt_kernel<<<(n_mlp + 255) / 256, 256, 0, stream>>>(w_mlp, w_mlp_bf, n_mlp);
  cvt_kernel<<<(n_inp + 255) / 256, 256, 0, stream>>>(inpw,  inpw_bf,  n_inp);
  cvt_kernel<<<(n_out + 255) / 256, 256, 0, stream>>>(outpw, outpw_bf, n_out);
  cvt_kernel<<<(n_ff1 + 255) / 256, 256, 0, stream>>>(ffw1,  ffw1_bf,  n_ff1);
  cvt_kernel<<<(n_ff2 + 255) / 256, 256, 0, stream>>>(ffw2,  ffw2_bf,  n_ff2);
  cvt_kernel<<<(n_cnv + 255) / 256, 256, 0, stream>>>(convw, convw_bf, n_cnv);

  const dim3 tg(LTOK / 16, BM);
  token_gemm_kernel<<<tg, 128, 0, stream>>>(buffer, w_mlp_bf, tokf);
  token_gemm_kernel<<<tg, 128, 0, stream>>>(spa,    w_mlp_bf, pef);

  ln1_kernel<<<MTOT / 8, 256, 0, stream>>>(tokf, pef, ln1g, ln1b, xbf, tokbf);

  linear_kernel<0><<<dim3(MTOT/64, 1), 128, 0, stream>>>(xbf,   inpw_bf,            MTOT, EDIM, EDIM, qb, nullptr, nullptr);
  linear_kernel<0><<<dim3(MTOT/64, 1), 128, 0, stream>>>(xbf,   inpw_bf + 128*128,  MTOT, EDIM, EDIM, kb, nullptr, nullptr);
  linear_kernel<0><<<dim3(MTOT/64, 1), 128, 0, stream>>>(tokbf, inpw_bf + 256*128,  MTOT, EDIM, EDIM, vb, nullptr, nullptr);

  attn_kernel<<<dim3(LTOK/16, NHEAD, BM), 32, 0, stream>>>(qb, kb, vb, xbf);

  linear_kernel<2><<<dim3(MTOT/64, 1), 128, 0, stream>>>(xbf, outpw_bf, MTOT, EDIM, EDIM, nullptr, pef, tokf);

  ln2_kernel<<<MTOT / 8, 256, 0, stream>>>(pef, ln2g, ln2b, tokbf);

  linear_kernel<1><<<dim3(MTOT/64, 2), 128, 0, stream>>>(tokbf, ffw1_bf, MTOT, 256, EDIM, qb, nullptr, nullptr);

  linear_kernel<3><<<dim3(MTOT/64, 1), 128, 0, stream>>>(qb, ffw2_bf, MTOT, EDIM, 256, vb, nullptr, pef);

  linear_kernel<4><<<dim3(MTOT/64, 1), 128, 0, stream>>>(vb, convw_bf, MTOT, 64, EDIM, nullptr, out, nullptr);
}
